// MoleculeFusionModel_50328426775225
// MI455X (gfx1250) — matrix operand order repaired and checked by reading, not compiled, not run on hardware
//
#include <hip/hip_runtime.h>
#include <hip/hip_bf16.h>

#define HEADS      3
#define NEG_SLOPE  0.2f
#define MAX_NODES  29
#define NUM_FEAT   11
#define HID        64
#define LAT        64
#define NUM_GRAPHS 2048

typedef __attribute__((ext_vector_type(16))) _Float16 v16h;
typedef __attribute__((ext_vector_type(8)))  _Float16 v8h;
typedef __attribute__((ext_vector_type(8)))  float    v8f;
typedef __attribute__((ext_vector_type(4)))  int      v4i;

// LDS tile row stride in halves: 32 data + 8 pad = 40 (80 B = 20 banks, 16B aligned)
#define TSTRIDE 40

#if defined(__has_builtin)
#if __has_builtin(__builtin_amdgcn_global_load_async_to_lds_b128) && \
    __has_builtin(__builtin_amdgcn_s_wait_asynccnt)
#define HAVE_ASYNC_LDS 1
#endif
#endif

// ---------- helpers ----------
__device__ __forceinline__ unsigned f2ord(float f) {
  unsigned u = __float_as_uint(f);
  return (u & 0x80000000u) ? ~u : (u | 0x80000000u);
}
__device__ __forceinline__ float ord2f(unsigned v) {
  unsigned u = (v & 0x80000000u) ? (v & 0x7FFFFFFFu) : ~v;
  return __uint_as_float(u);
}

// two contiguous 16B runs -> one 16-half fragment
__device__ __forceinline__ v16h load_frag(const _Float16* p, int off2) {
  v8h lo = *(const v8h*)p;
  v8h hi = *(const v8h*)(p + off2);
  v16h r;
#pragma unroll
  for (int j = 0; j < 8; ++j) { r[j] = lo[j]; r[j + 8] = hi[j]; }
  return r;
}

__global__ __launch_bounds__(256)
void zero_b32(unsigned* __restrict__ p, long n) {
  long i = (long)blockIdx.x * blockDim.x + threadIdx.x;
  if (i < n) p[i] = 0u;
}

// ---------- shared GEMM pieces ----------
// stage B tile (32 k x 64 cols) from fp32 W, transposed+converted into LDS
__device__ __forceinline__ void stage_B(const float* __restrict__ W,
                                        _Float16* __restrict__ Bs,
                                        int t, int kk, int K, int M, int col0) {
  if ((col0 + 64 <= M) && (kk + 32 <= K) && ((M & 3) == 0)) {
#pragma unroll
    for (int i = 0; i < 2; ++i) {
      int k  = i * 16 + (t >> 4);
      int cq = (t & 15) << 2;
      const float4 v = *(const float4*)(W + (long)(kk + k) * M + col0 + cq);
      Bs[(cq + 0) * TSTRIDE + k] = (_Float16)v.x;
      Bs[(cq + 1) * TSTRIDE + k] = (_Float16)v.y;
      Bs[(cq + 2) * TSTRIDE + k] = (_Float16)v.z;
      Bs[(cq + 3) * TSTRIDE + k] = (_Float16)v.w;
    }
  } else {
#pragma unroll
    for (int i = 0; i < 2; ++i) {
      int kl = i * 16 + (t >> 4);
      int k  = kk + kl;
      int cq = (t & 15) << 2;
#pragma unroll
      for (int c = 0; c < 4; ++c) {
        int col = col0 + cq + c;
        float f = (k < K && col < M) ? W[(long)k * M + col] : 0.0f;
        Bs[(cq + c) * TSTRIDE + kl] = (_Float16)f;
      }
    }
  }
}

// ---------- WMMA GEMM (fp32 A): C[N,M] = act(A[N,K] @ W[K,M] + bias[M]) ----------
// 256 threads = 8 waves, 64x64 tile, K-step 32, LDS-staged f16 fragments.
__global__ __launch_bounds__(256)
void wmma_gemm(const float* __restrict__ A, const float* __restrict__ W,
               const float* __restrict__ bias, float* __restrict__ C,
               int N, int K, int M, int relu) {
  __shared__ _Float16 As[64 * TSTRIDE];   // row-major rows x k
  __shared__ _Float16 Bs[64 * TSTRIDE];   // transposed: col x k

  const int t      = threadIdx.x;
  const int lane   = t & 31;
  const int wave   = t >> 5;
  const int laneHi = (lane >= 16) ? 1 : 0;
  const int l15    = lane & 15;
  const int row0   = blockIdx.x * 64;
  const int col0   = blockIdx.y * 64;
  const int rs     = wave >> 1;   // row strip 0..3
  const int ch     = wave & 1;    // col half 0..1

  v8f acc0 = {};
  v8f acc1 = {};

  for (int kk = 0; kk < K; kk += 32) {
    __syncthreads();
    if (kk + 32 < K) {  // prefetch next W tile (speculative, dropped if invalid)
      __builtin_prefetch((const void*)(W + (long)(kk + 32 + (t >> 4)) * M +
                                       col0 + ((t & 15) << 2)), 0, 3);
    }
    // ---- stage A tile (64 rows x 32 k) ----
    if ((row0 + 64 <= N) && (kk + 32 <= K) && ((K & 3) == 0)) {
#pragma unroll
      for (int i = 0; i < 2; ++i) {
        int r  = i * 32 + (t >> 3);
        int kq = (t & 7) << 2;
        const float4 v = *(const float4*)(A + (long)(row0 + r) * K + kk + kq);
        _Float16* d = &As[r * TSTRIDE + kq];
        d[0] = (_Float16)v.x; d[1] = (_Float16)v.y;
        d[2] = (_Float16)v.z; d[3] = (_Float16)v.w;
      }
    } else {
#pragma unroll
      for (int i = 0; i < 2; ++i) {
        int r    = i * 32 + (t >> 3);
        int kq   = (t & 7) << 2;
        int grow = row0 + r;
        _Float16* d = &As[r * TSTRIDE + kq];
#pragma unroll
        for (int c = 0; c < 4; ++c) {
          int k = kk + kq + c;
          float f = (grow < N && k < K) ? A[(long)grow * K + k] : 0.0f;
          d[c] = (_Float16)f;
        }
      }
    }
    stage_B(W, Bs, t, kk, K, M, col0);
    __syncthreads();

    // ---- fragments (2x ds_load_b128 each) + WMMA ----
    const int kbA = laneHi ? 8 : 0;
    const int kbB = laneHi ? 8 : 0;
    v16h a  = load_frag(&As[(rs * 16 + l15) * TSTRIDE + kbA], 16);
    v16h b0 = load_frag(&Bs[((ch * 2 + 0) * 16 + l15) * TSTRIDE + kbB], 16);
    v16h b1 = load_frag(&Bs[((ch * 2 + 1) * 16 + l15) * TSTRIDE + kbB], 16);
    acc0 = __builtin_amdgcn_wmma_f32_16x16x32_f16(false, a, false, b0,
                                                  (short)0, acc0, false, false);
    acc1 = __builtin_amdgcn_wmma_f32_16x16x32_f16(false, a, false, b1,
                                                  (short)0, acc1, false, false);
  }

#pragma unroll
  for (int r = 0; r < 8; ++r) {
    int m  = row0 + rs * 16 + r + (laneHi ? 8 : 0);
    int n0 = col0 + (ch * 2 + 0) * 16 + l15;
    int n1 = col0 + (ch * 2 + 1) * 16 + l15;
    if (m < N && n0 < M) {
      float v = acc0[r] + bias[n0];
      if (relu) v = fmaxf(v, 0.0f);
      C[(long)m * M + n0] = v;
    }
    if (m < N && n1 < M) {
      float v = acc1[r] + bias[n1];
      if (relu) v = fmaxf(v, 0.0f);
      C[(long)m * M + n1] = v;
    }
  }
}

// ---------- WMMA GEMM (f16 A mirror): async global->LDS staging of A ----------
__global__ __launch_bounds__(256)
void wmma_gemm_h(const _Float16* __restrict__ A16, const float* __restrict__ W,
                 const float* __restrict__ bias, float* __restrict__ C,
                 int N, int K, int M, int relu) {
  __shared__ _Float16 As[64 * TSTRIDE];
  __shared__ _Float16 Bs[64 * TSTRIDE];

  const int t      = threadIdx.x;
  const int lane   = t & 31;
  const int wave   = t >> 5;
  const int laneHi = (lane >= 16) ? 1 : 0;
  const int l15    = lane & 15;
  const int row0   = blockIdx.x * 64;
  const int col0   = blockIdx.y * 64;
  const int rs     = wave >> 1;
  const int ch     = wave & 1;

  v8f acc0 = {};
  v8f acc1 = {};

  for (int kk = 0; kk < K; kk += 32) {
    __syncthreads();
    if (kk + 32 < K) {
      __builtin_prefetch((const void*)(W + (long)(kk + 32 + (t >> 4)) * M +
                                       col0 + ((t & 15) << 2)), 0, 3);
    }
    // ---- stage A tile (64 rows x 32 k, f16): one 16B segment per thread ----
    {
      const int r = t >> 2;           // 0..63
      const int q = t & 3;            // 16B quarter of the 64B row segment
      const bool fast = (row0 + 64 <= N) && (kk + 32 <= K) && ((K & 7) == 0);
      if (fast) {
#if defined(HAVE_ASYNC_LDS)
        const _Float16* g = A16 + (long)(row0 + r) * K + kk + q * 8;
        __builtin_amdgcn_global_load_async_to_lds_b128(
            (__attribute__((address_space(1))) v4i*)(void*)g,
            (__attribute__((address_space(3))) v4i*)(void*)&As[r * TSTRIDE + q * 8],
            0, 0);
#else
        v8h v = *(const v8h*)(A16 + (long)(row0 + r) * K + kk + q * 8);
        *(v8h*)&As[r * TSTRIDE + q * 8] = v;
#endif
      } else {
        int grow = row0 + r;
#pragma unroll
        for (int j = 0; j < 8; ++j) {
          int k = kk + q * 8 + j;
          As[r * TSTRIDE + q * 8 + j] =
              (grow < N && k < K) ? A16[(long)grow * K + k] : (_Float16)0.0f;
        }
      }
    }
    stage_B(W, Bs, t, kk, K, M, col0);
#if defined(HAVE_ASYNC_LDS)
    __builtin_amdgcn_s_wait_asynccnt(0);
#endif
    __syncthreads();

    const int kbA = laneHi ? 8 : 0;
    const int kbB = laneHi ? 8 : 0;
    v16h a  = load_frag(&As[(rs * 16 + l15) * TSTRIDE + kbA], 16);
    v16h b0 = load_frag(&Bs[((ch * 2 + 0) * 16 + l15) * TSTRIDE + kbB], 16);
    v16h b1 = load_frag(&Bs[((ch * 2 + 1) * 16 + l15) * TSTRIDE + kbB], 16);
    acc0 = __builtin_amdgcn_wmma_f32_16x16x32_f16(false, a, false, b0,
                                                  (short)0, acc0, false, false);
    acc1 = __builtin_amdgcn_wmma_f32_16x16x32_f16(false, a, false, b1,
                                                  (short)0, acc1, false, false);
  }

#pragma unroll
  for (int r = 0; r < 8; ++r) {
    int m  = row0 + rs * 16 + r + (laneHi ? 8 : 0);
    int n0 = col0 + (ch * 2 + 0) * 16 + l15;
    int n1 = col0 + (ch * 2 + 1) * 16 + l15;
    if (m < N && n0 < M) {
      float v = acc0[r] + bias[n0];
      if (relu) v = fmaxf(v, 0.0f);
      C[(long)m * M + n0] = v;
    }
    if (m < N && n1 < M) {
      float v = acc1[r] + bias[n1];
      if (relu) v = fmaxf(v, 0.0f);
      C[(long)m * M + n1] = v;
    }
  }
}

// ---------- GATv2 edge kernels (self-loops synthesized: e>=E -> src=dst=e-E) ----------
__global__ __launch_bounds__(256)
void edge_logits_kernel(const float* __restrict__ xl, const float* __restrict__ xr,
                        const float* __restrict__ att, const int* __restrict__ ei,
                        int E, int Etot, int C, float* __restrict__ ework,
                        unsigned* __restrict__ mx) {
  const int lane = threadIdx.x & 31;
  long e = (long)blockIdx.x * (blockDim.x >> 5) + (threadIdx.x >> 5);
  if (e >= Etot) return;
  int src = (e < E) ? ei[e]           : (int)(e - E);
  int dst = (e < E) ? ei[(long)E + e] : (int)(e - E);
  const float* pl = xl + (long)src * (HEADS * C);
  const float* pr = xr + (long)dst * (HEADS * C);
#pragma unroll
  for (int h = 0; h < HEADS; ++h) {
    float acc = 0.0f;
    for (int c = lane; c < C; c += 32) {
      float v = pl[h * C + c] + pr[h * C + c];
      v = (v > 0.0f) ? v : NEG_SLOPE * v;          // leaky_relu
      acc = fmaf(v, att[h * C + c], acc);
    }
#pragma unroll
    for (int o = 16; o > 0; o >>= 1) acc += __shfl_down(acc, o, 32);
    if (lane == 0) {
      ework[e * HEADS + h] = acc;
      atomicMax(&mx[(long)dst * HEADS + h], f2ord(acc));   // segment max
    }
  }
}

__global__ __launch_bounds__(256)
void edge_exp_kernel(float* __restrict__ ework, const unsigned* __restrict__ mx,
                     float* __restrict__ denom, const int* __restrict__ ei,
                     int E, int Etot) {
  long idx = (long)blockIdx.x * blockDim.x + threadIdx.x;
  long e = idx / HEADS;
  int  h = (int)(idx % HEADS);
  if (e >= Etot) return;
  int dst = (e < E) ? ei[(long)E + e] : (int)(e - E);
  float a = expf(ework[e * HEADS + h] - ord2f(mx[(long)dst * HEADS + h]));
  ework[e * HEADS + h] = a;
  atomicAdd(&denom[(long)dst * HEADS + h], a);         // segment sum
}

__global__ __launch_bounds__(256)
void edge_scatter_kernel(const float* __restrict__ ework, const float* __restrict__ denom,
                         const float* __restrict__ xl, const int* __restrict__ ei,
                         int E, int Etot, int C, float* __restrict__ hout) {
  long idx = (long)blockIdx.x * blockDim.x + threadIdx.x;
  long e = idx / C;
  int  c = (int)(idx % C);
  if (e >= Etot) return;
  int src = (e < E) ? ei[e]           : (int)(e - E);
  int dst = (e < E) ? ei[(long)E + e] : (int)(e - E);
  float s = 0.0f;
#pragma unroll
  for (int h = 0; h < HEADS; ++h) {
    float alpha = ework[e * HEADS + h] / (denom[(long)dst * HEADS + h] + 1e-16f);
    s = fmaf(alpha, xl[(long)src * (HEADS * C) + (long)h * C + c], s);
  }
  atomicAdd(&hout[(long)dst * C + c], s * (1.0f / (float)HEADS));  // mean over heads
}

// bias + relu; optionally emit f16 mirror for the next layer's WMMA A-matrix
__global__ __launch_bounds__(256)
void bias_relu_kernel(float* __restrict__ h, const float* __restrict__ bias,
                      _Float16* __restrict__ mirror, long n, int C) {
  long idx = (long)blockIdx.x * blockDim.x + threadIdx.x;
  if (idx >= n) return;
  int c = (int)(idx % C);
  float v = fmaxf(h[idx] + bias[c], 0.0f);
  h[idx] = v;
  if (mirror) mirror[idx] = (_Float16)v;
}

__global__ __launch_bounds__(256)
void pool_kernel(const float* __restrict__ h, const int* __restrict__ batch,
                 float* __restrict__ g, int N, int C) {
  long idx = (long)blockIdx.x * blockDim.x + threadIdx.x;
  long node = idx / C;
  int  c    = (int)(idx % C);
  if (node >= N) return;
  atomicAdd(&g[(long)batch[node] * C + c], h[(long)node * C + c]);
}

// split o3 -> x_rec and symmetrized adj
__global__ __launch_bounds__(256)
void finalize_kernel(const float* __restrict__ o3, float* __restrict__ xrec,
                     float* __restrict__ adj) {
  const int OD = MAX_NODES * NUM_FEAT + MAX_NODES * MAX_NODES; // 1160
  const int NS = MAX_NODES * NUM_FEAT;                          // 319
  long idx = (long)blockIdx.x * blockDim.x + threadIdx.x;
  if (idx >= (long)NUM_GRAPHS * OD) return;
  int gI  = (int)(idx / OD);
  int col = (int)(idx % OD);
  if (col < NS) {
    xrec[(long)gI * NS + col] = o3[idx];
  } else {
    int q = col - NS;
    int i = q / MAX_NODES, j = q % MAX_NODES;
    const float* base = o3 + (long)gI * OD + NS;
    adj[(long)gI * (MAX_NODES * MAX_NODES) + q] =
        0.5f * (base[i * MAX_NODES + j] + base[j * MAX_NODES + i]);
  }
}

// ---------- host ----------
extern "C" void kernel_launch(void* const* d_in, const int* in_sizes, int n_in,
                              void* d_out, int out_size, void* d_ws, size_t ws_size,
                              hipStream_t stream) {
  (void)n_in; (void)out_size; (void)ws_size;
  const float* x     = (const float*)d_in[0];
  const int*   ei    = (const int*)d_in[1];
  const int*   batch = (const int*)d_in[2];
  const float *Wl1=(const float*)d_in[3],  *bl1=(const float*)d_in[4];
  const float *Wr1=(const float*)d_in[5],  *br1=(const float*)d_in[6];
  const float *att1=(const float*)d_in[7], *b1=(const float*)d_in[8];
  const float *Wl2=(const float*)d_in[9],  *bl2=(const float*)d_in[10];
  const float *Wr2=(const float*)d_in[11], *br2=(const float*)d_in[12];
  const float *att2=(const float*)d_in[13],*b2=(const float*)d_in[14];
  const float *Wl3=(const float*)d_in[15], *bl3=(const float*)d_in[16];
  const float *Wr3=(const float*)d_in[17], *br3=(const float*)d_in[18];
  const float *att3=(const float*)d_in[19],*b3=(const float*)d_in[20];
  const float *Wmu=(const float*)d_in[21], *bmu=(const float*)d_in[22];
  const float *Wlv=(const float*)d_in[23], *blv=(const float*)d_in[24];
  const float *Wd1=(const float*)d_in[25], *bd1=(const float*)d_in[26];
  const float *Wd2=(const float*)d_in[27], *bd2=(const float*)d_in[28];
  const float *Wd3=(const float*)d_in[29], *bd3=(const float*)d_in[30];

  const int N    = in_sizes[2];       // 50000 nodes
  const int E    = in_sizes[1] / 2;   // 400000 edges
  const int Etot = E + N;             // + self loops
  const int OUTD = MAX_NODES * NUM_FEAT + MAX_NODES * MAX_NODES; // 1160

  // output layout: x_rec | adj_rec | mu | logvar
  float* out = (float*)d_out;
  const size_t nx   = (size_t)NUM_GRAPHS * MAX_NODES * NUM_FEAT;
  const size_t nadj = (size_t)NUM_GRAPHS * MAX_NODES * MAX_NODES;
  float* muP = out + nx + nadj;
  float* lvP = muP + (size_t)NUM_GRAPHS * LAT;

  // workspace carve (256B aligned)
  size_t off = 0;
  auto carveB = [&](size_t bytes) -> void* {
    void* p = (char*)d_ws + off;
    off += (bytes + 255) & ~(size_t)255;
    return p;
  };
  float*    xl    = (float*)carveB((size_t)N * (HEADS * 256) * 4);
  float*    xr    = (float*)carveB((size_t)N * (HEADS * 256) * 4);
  float*    hA    = (float*)carveB((size_t)N * 256 * 4);
  float*    hB    = (float*)carveB((size_t)N * 256 * 4);
  _Float16* hA16  = (_Float16*)carveB((size_t)N * HID * 2);        // L2 A-matrix
  _Float16* hB16  = (_Float16*)carveB((size_t)N * 2 * HID * 2);    // L3 A-matrix
  float*    ework = (float*)carveB((size_t)Etot * HEADS * 4);
  unsigned* mx    = (unsigned*)carveB((size_t)N * HEADS * 4);
  float*    denom = (float*)carveB((size_t)N * HEADS * 4);
  float*    gbuf  = (float*)carveB((size_t)NUM_GRAPHS * 256 * 4);
  float*    o1    = (float*)carveB((size_t)NUM_GRAPHS * 256 * 4);
  float*    o2    = (float*)carveB((size_t)NUM_GRAPHS * 512 * 4);
  float*    o3    = (float*)carveB((size_t)NUM_GRAPHS * OUTD * 4);

  auto zeroN = [&](void* p, size_t n) {
    zero_b32<<<(unsigned)((n + 255) / 256), 256, 0, stream>>>((unsigned*)p, (long)n);
  };
  auto gemm = [&](const float* A, const float* W, const float* b, float* Cp,
                  int n, int k, int m, int relu) {
    dim3 grid((n + 63) / 64, (m + 63) / 64);
    wmma_gemm<<<grid, dim3(256), 0, stream>>>(A, W, b, Cp, n, k, m, relu);
  };
  auto gemmh = [&](const _Float16* A16, const float* W, const float* b, float* Cp,
                   int n, int k, int m, int relu) {
    dim3 grid((n + 63) / 64, (m + 63) / 64);
    wmma_gemm_h<<<grid, dim3(256), 0, stream>>>(A16, W, b, Cp, n, k, m, relu);
  };
  auto attn = [&](const float* xlp, const float* xrp, const float* att,
                  const float* bias, float* hout, _Float16* mirror, int C) {
    zeroN(mx,    (size_t)N * HEADS);
    zeroN(denom, (size_t)N * HEADS);
    zeroN(hout,  (size_t)N * C);
    long nwaves = (long)Etot;                       // one wave per edge
    edge_logits_kernel<<<(unsigned)((nwaves + 7) / 8), 256, 0, stream>>>(
        xlp, xrp, att, ei, E, Etot, C, ework, mx);
    long t1 = (long)Etot * HEADS;
    edge_exp_kernel<<<(unsigned)((t1 + 255) / 256), 256, 0, stream>>>(
        ework, mx, denom, ei, E, Etot);
    long t2 = (long)Etot * C;
    edge_scatter_kernel<<<(unsigned)((t2 + 255) / 256), 256, 0, stream>>>(
        ework, denom, xlp, ei, E, Etot, C, hout);
    long t3 = (long)N * C;
    bias_relu_kernel<<<(unsigned)((t3 + 255) / 256), 256, 0, stream>>>(
        hout, bias, mirror, t3, C);
  };

  // ---- GATv2 layer 1: 11 -> 64 (heads*64 = 192) ----
  gemm(x, Wl1, bl1, xl, N, NUM_FEAT, HEADS * HID, 0);
  gemm(x, Wr1, br1, xr, N, NUM_FEAT, HEADS * HID, 0);
  attn(xl, xr, att1, b1, hA, hA16, HID);

  // ---- GATv2 layer 2: 64 -> 128 (heads*128 = 384) ----
  gemmh(hA16, Wl2, bl2, xl, N, HID, HEADS * 2 * HID, 0);
  gemmh(hA16, Wr2, br2, xr, N, HID, HEADS * 2 * HID, 0);
  attn(xl, xr, att2, b2, hB, hB16, 2 * HID);

  // ---- GATv2 layer 3: 128 -> 256 (heads*256 = 768) ----
  gemmh(hB16, Wl3, bl3, xl, N, 2 * HID, HEADS * 4 * HID, 0);
  gemmh(hB16, Wr3, br3, xr, N, 2 * HID, HEADS * 4 * HID, 0);
  attn(xl, xr, att3, b3, hA, (_Float16*)nullptr, 4 * HID);

  // ---- global_add_pool ----
  zeroN(gbuf, (size_t)NUM_GRAPHS * 4 * HID);
  long tp = (long)N * (4 * HID);
  pool_kernel<<<(unsigned)((tp + 255) / 256), 256, 0, stream>>>(hA, batch, gbuf, N, 4 * HID);

  // ---- mu / logvar ----
  gemm(gbuf, Wmu, bmu, muP, NUM_GRAPHS, 4 * HID, LAT, 0);
  gemm(gbuf, Wlv, blv, lvP, NUM_GRAPHS, 4 * HID, LAT, 0);

  // ---- decoder (z = mu) ----
  gemm(muP, Wd1, bd1, o1, NUM_GRAPHS, LAT,     4 * HID, 1);
  gemm(o1,  Wd2, bd2, o2, NUM_GRAPHS, 4 * HID, 8 * HID, 1);
  gemm(o2,  Wd3, bd3, o3, NUM_GRAPHS, 8 * HID, OUTD,    0);

  // ---- split + symmetrize ----
  long tf = (long)NUM_GRAPHS * OUTD;
  finalize_kernel<<<(unsigned)((tf + 255) / 256), 256, 0, stream>>>(o3, out, out + nx);
}
